// LoRALinear_25984552141356
// MI455X (gfx1250) — matrix operand order repaired and checked by reading, not compiled, not run on hardware
//
#include <hip/hip_runtime.h>

// Plain vector types (union-safe, avoid HIP's class-based vector types)
typedef __attribute__((ext_vector_type(16))) _Float16     v16h;
typedef __attribute__((ext_vector_type(8)))  float        v8f;
typedef __attribute__((ext_vector_type(4)))  unsigned int v4u;
typedef __attribute__((ext_vector_type(2)))  float        v2f;

#define TOKENS 16384
#define IN_F   4096
#define OUT_F  4096
#define RANK   16
#define SCALING 2.0f

#define BM 128
#define BN 128
#define BK 64
#define KT (IN_F / 32)   // 128 K-fragments of 32 along the reduction dim

#if __has_builtin(__builtin_amdgcn_global_load_async_to_lds_b128) && \
    __has_builtin(__builtin_amdgcn_s_wait_asynccnt)
#define USE_ASYNC_B 1
// Builtin signature (from compiler diagnostic): param is a pointer to a
// 16-byte int vector; global operand in AS1, LDS operand in AS3.
typedef int gvec4i __attribute__((vector_size(4 * sizeof(int))));
typedef __attribute__((address_space(1))) gvec4i GV4;
typedef __attribute__((address_space(3))) gvec4i LV4;
#else
#define USE_ASYNC_B 0
#endif

// ---------------------------------------------------------------------------
// P1: per-output-row symmetric int8 scale = absmax / 127
// ---------------------------------------------------------------------------
__global__ __launch_bounds__(256) void k_rowscale(const float* __restrict__ w,
                                                  float* __restrict__ scale)
{
    __shared__ float red[256];
    const int o = blockIdx.x;
    const float* row = w + (size_t)o * IN_F;
    float m = 0.0f;
    for (int i = threadIdx.x; i < IN_F; i += 256)
        m = fmaxf(m, fabsf(row[i]));
    red[threadIdx.x] = m;
    __syncthreads();
    for (int s = 128; s > 0; s >>= 1) {
        if (threadIdx.x < s)
            red[threadIdx.x] = fmaxf(red[threadIdx.x], red[threadIdx.x + s]);
        __syncthreads();
    }
    if (threadIdx.x == 0) scale[o] = red[0] / 127.0f;
}

// ---------------------------------------------------------------------------
// P2: M[k][o] = dequant(W[o][k]) + 2 * (lora_a @ lora_b)[k][o], stored f16,
// pre-swizzled into B fragments for 16x16x32_f16 in this file's contiguous K order:
//   fragment (ot, kt) = 1 KB: 32 lanes x 32 B; lane%16 = column N, lanes 0-15
//   hold K=0..15 (K=2v+h in VGPR v), lanes 16-31 hold K=16..31.
// ---------------------------------------------------------------------------
__global__ __launch_bounds__(256) void k_pack(const float* __restrict__ w,
                                              const float* __restrict__ la,
                                              const float* __restrict__ lb,
                                              const float* __restrict__ scale,
                                              unsigned int* __restrict__ packed)
{
    const int ot   = blockIdx.x;          // 0..255 (16 output cols each)
    const int ktg  = blockIdx.y;          // 0..7   (16 kt fragments each)
    const int lane = threadIdx.x >> 3;    // 0..31
    const int v    = threadIdx.x & 7;     // 0..7 (VGPR slot)
    const int o    = ot * 16 + (lane & 15);
    const int kk   = ((lane < 16) ? 0 : 16) + v * 2;  // B-layout K within 32
    const float s   = scale[o];
    const float inv = 1.0f / (s + 1e-8f);

    float bcol[RANK];
#pragma unroll
    for (int r = 0; r < RANK; ++r) bcol[r] = lb[r * OUT_F + o];

    for (int i = 0; i < 16; ++i) {
        const int kt = ktg * 16 + i;
        const int k  = kt * 32 + kk;
        const float w0 = w[(size_t)o * IN_F + k];
        const float w1 = w[(size_t)o * IN_F + k + 1];
        // round-half-even (rintf) + clip, matching jnp.round/jnp.clip
        const float q0 = fminf(fmaxf(rintf(w0 * inv), -128.0f), 127.0f);
        const float q1 = fminf(fmaxf(rintf(w1 * inv), -128.0f), 127.0f);
        float m0 = q0 * s, m1 = q1 * s;
        const float* a0 = la + (size_t)k * RANK;
        const float* a1 = a0 + RANK;
        float d0 = 0.0f, d1 = 0.0f;
#pragma unroll
        for (int r = 0; r < RANK; ++r) {
            d0 = fmaf(a0[r], bcol[r], d0);
            d1 = fmaf(a1[r], bcol[r], d1);
        }
        m0 = fmaf(SCALING, d0, m0);
        m1 = fmaf(SCALING, d1, m1);
        union { _Float16 h[2]; unsigned int u; } pk;
        pk.h[0] = (_Float16)m0;
        pk.h[1] = (_Float16)m1;
        packed[(((size_t)ot * KT + kt) * 32 + lane) * 8 + v] = pk.u;
    }
}

// ---------------------------------------------------------------------------
// Main GEMM: out[T][OUT] = x[T][IN] @ M[IN][OUT] via v_wmma_f32_16x16x32_f16.
// 256 threads = 8 waves; block tile 128x128, BK=64 (2 WMMA K-steps/stage).
// A tile: f32->f16 VALU convert into LDS (double-buffered).
// B tile: async global->LDS copy (ASYNCcnt), double-buffered; compute reads
// both operands with ds_load_b128, so WMMAs don't wait on global latency.
// ---------------------------------------------------------------------------
__global__ __launch_bounds__(256) void k_gemm(const float* __restrict__ x,
                                              const v4u* __restrict__ Bp,
                                              float* __restrict__ out)
{
    __shared__ __attribute__((aligned(32))) unsigned int ldsA[2][BM * BK / 2]; // 2 x 16 KB
#if USE_ASYNC_B
    __shared__ __attribute__((aligned(32))) unsigned int ldsB[2][BK * BN / 2]; // 2 x 16 KB
#endif

    const int tid  = threadIdx.x;
    const int wave = tid >> 5, lane = tid & 31;
    const int wr   = wave >> 1, wc = wave & 1;
    const int bm   = blockIdx.y * BM;
    const int bn   = blockIdx.x * BN;
    const int ot0  = bn >> 4;

    // ---- A staging: 128x64 f32 -> f16 in WMMA A-fragment layout ----
    // A 16x32 f16 order used here (contiguous K, same as B): lanes 0-15 row m, K in {0..15};
    // lanes 16-31 K in {16..31}; VGPR v holds a consecutive K pair.
    auto stageA = [&](int k0, int buf) {
#pragma unroll
        for (int p = 0; p < 16; ++p) {
            const int m  = p * 8 + wave;     // 0..127
            const int kc = lane * 2;         // 0..62 (even)
            const float* src = x + (size_t)(bm + m) * IN_F + k0 + kc;
            const v2f a = *(const v2f*)src;
            if (p == 0 && k0 + BK < IN_F)
                __builtin_prefetch(src + BK, 0, 0);   // global_prefetch_b8
            const int rt = m >> 4, m16 = m & 15;
            const int ks = kc >> 5, kkk = kc & 31;
            const int al = m16 + ((kkk >= 16) ? 16 : 0);
            const int av = (kkk & 15) >> 1;
            union { _Float16 h[2]; unsigned int u; } pk;
            pk.h[0] = (_Float16)a.x;
            pk.h[1] = (_Float16)a.y;
            ldsA[buf][(((rt * 2 + ks) * 32 + al) * 8) + av] = pk.u;
        }
    };

#if USE_ASYNC_B
    // ---- B staging: async copy 16 fragments (8 ot x 2 ks) = 16 KB ----
    // Fragment f = otLocal*2 + ks, stored verbatim (already fragment layout).
    auto stageB = [&](int ktb, int buf) {
#pragma unroll
        for (int q = 0; q < 4; ++q) {
            const int flat  = q * 256 + tid;        // 0..1023 (16B units)
            const int frag  = flat >> 6;            // 0..15
            const int inner = (flat & 63) * 16;     // byte offset in fragment
            const int otl   = frag >> 1, ks = frag & 1;
            const char* src = (const char*)Bp +
                ((((size_t)(ot0 + otl) * KT) + (ktb + ks)) << 10) + inner;
            unsigned int* dst = &ldsB[buf][frag * 256 + (flat & 63) * 4];
            __builtin_amdgcn_global_load_async_to_lds_b128(
                (GV4*)src, (LV4*)dst, 0, 0);
        }
    };
#endif

    const v8f vzero = {0.f, 0.f, 0.f, 0.f, 0.f, 0.f, 0.f, 0.f};
    v8f acc[2][4];
#pragma unroll
    for (int i = 0; i < 2; ++i)
#pragma unroll
        for (int j = 0; j < 4; ++j) acc[i][j] = vzero;

#if !USE_ASYNC_B
    // Per-lane base for this wave's B fragments (register path fallback)
    const v4u* bbase = Bp + (((size_t)(ot0 + wc * 4) * KT) * 32 + lane) * 2;
#endif

    stageA(0, 0);
#if USE_ASYNC_B
    stageB(0, 0);
#endif
    int cur = 0;
    for (int k0 = 0; k0 < IN_F; k0 += BK) {
#if USE_ASYNC_B
        __builtin_amdgcn_s_wait_asynccnt(0);   // own async B writes landed
#endif
        __syncthreads();                       // buffers `cur` fully staged
        const int ktb = k0 >> 5;
        if (k0 + BK < IN_F) {
            stageA(k0 + BK, cur ^ 1);          // overlap next stage with compute
#if USE_ASYNC_B
            stageB(ktb + 2, cur ^ 1);
#endif
        }

#pragma unroll
        for (int ks = 0; ks < 2; ++ks) {
            // A fragments for this K-step
            v16h aF[2];
#pragma unroll
            for (int rt = 0; rt < 2; ++rt)
                aF[rt] = *(const v16h*)(&ldsA[cur][(((wr * 2 + rt) * 2 + ks) * 32 + lane) * 8]);
            // B fragments for this K-step
            union { v4u u[2]; v16h h; } bF[4];
#pragma unroll
            for (int ct = 0; ct < 4; ++ct) {
#if USE_ASYNC_B
                const unsigned int* bp =
                    &ldsB[cur][(((wc * 4 + ct) * 2 + ks) * 256) + lane * 8];
                bF[ct].u[0] = *(const v4u*)bp;
                bF[ct].u[1] = *(const v4u*)(bp + 4);
#else
                const v4u* bp = bbase + ((size_t)ct * KT + (ktb + ks)) * 64;
                bF[ct].u[0] = bp[0];
                bF[ct].u[1] = bp[1];
#endif
            }
#pragma unroll
            for (int ct = 0; ct < 4; ++ct)
#pragma unroll
                for (int rt = 0; rt < 2; ++rt)
                    acc[rt][ct] = __builtin_amdgcn_wmma_f32_16x16x32_f16(
                        false, aF[rt], false, bF[ct].h, (short)0,
                        acc[rt][ct], false, false);
        }
        cur ^= 1;
    }

    // ---- store: C/D layout — lane n = l%16, VGPR r -> row r + (l<16?0:8) ----
#pragma unroll
    for (int rt = 0; rt < 2; ++rt) {
        const int row_base = bm + (wr * 2 + rt) * 16 + ((lane < 16) ? 0 : 8);
#pragma unroll
        for (int ct = 0; ct < 4; ++ct) {
            const int col = bn + (wc * 4 + ct) * 16 + (lane & 15);
#pragma unroll
            for (int r = 0; r < 8; ++r)
                out[(size_t)(row_base + r) * OUT_F + col] = acc[rt][ct][r];
        }
    }
}

// ---------------------------------------------------------------------------
extern "C" void kernel_launch(void* const* d_in, const int* in_sizes, int n_in,
                              void* d_out, int out_size, void* d_ws, size_t ws_size,
                              hipStream_t stream)
{
    const float* x  = (const float*)d_in[0];   // [16384, 4096]
    const float* w  = (const float*)d_in[1];   // [4096, 4096]
    const float* la = (const float*)d_in[2];   // [4096, 16]
    const float* lb = (const float*)d_in[3];   // [16, 4096]
    float* out = (float*)d_out;                // [16384, 4096]

    // workspace: 32 MB packed f16 fused matrix + 16 KB scales
    unsigned int* packed = (unsigned int*)d_ws;
    float* scale = (float*)((char*)d_ws + (size_t)OUT_F * IN_F * sizeof(_Float16));

    k_rowscale<<<OUT_F, 256, 0, stream>>>(w, scale);
    k_pack<<<dim3(OUT_F / 16, KT / 16), 256, 0, stream>>>(w, la, lb, scale, packed);
    k_gemm<<<dim3(OUT_F / BN, TOKENS / BM), 256, 0, stream>>>(x, (const v4u*)packed, out);
}
